// DSRA_Chunk_Layer_35416300322985
// MI455X (gfx1250) — matrix operand order repaired and checked by reading, not compiled, not run on hardware
//
#include <hip/hip_runtime.h>
#include <hip/hip_bf16.h>

// ---------------------------------------------------------------------------
// Problem constants (match the reference)
// ---------------------------------------------------------------------------
constexpr int cB = 4, cT = 2048, cD = 1024, cK = 512, cKR = 16;
constexpr float cLAM = 0.01f, cETA = 0.1f;

constexpr size_t nBT  = (size_t)cB * cT;        // 8192 rows
constexpr size_t nBTD = nBT * cD;               // 8,388,608
constexpr size_t nBTK = nBT * cK;
constexpr size_t nBTT = (size_t)cB * cT * cT;
constexpr size_t nDD  = (size_t)cD * cD;
constexpr size_t nKD  = (size_t)cK * cD;
constexpr size_t nKK  = (size_t)cK * cK;
constexpr size_t nBKD = (size_t)cB * nKD;
constexpr size_t nBKK = (size_t)cB * nKK;

typedef __attribute__((ext_vector_type(16))) __bf16 v16bf;
typedef __attribute__((ext_vector_type(8)))  __bf16 v8bf;
typedef __attribute__((ext_vector_type(8)))  float  v8f;

// ---------------------------------------------------------------------------
// Wave32 helpers
// ---------------------------------------------------------------------------
__device__ inline float wave_sum(float v) {
#pragma unroll
  for (int o = 16; o > 0; o >>= 1) v += __shfl_xor(v, o, 32);
  return v;
}
__device__ inline float wave_max(float v) {
#pragma unroll
  for (int o = 16; o > 0; o >>= 1) v = fmaxf(v, __shfl_xor(v, o, 32));
  return v;
}

// Map each wave of a 256-thread block to one 32x64 macro-tile.
__device__ inline bool tile_coords(int M, int N, int& row0, int& col0) {
  int wave = threadIdx.x >> 5;
  int tile = blockIdx.x * (blockDim.x >> 5) + wave;
  int ntN  = N >> 6;
  int tm = tile / ntN, tn = tile - tm * ntN;
  if (tm >= (M >> 5)) return false;
  row0 = tm << 5; col0 = tn << 6;
  return true;
}

__device__ inline v16bf cat8(v8bf lo, v8bf hi) {
  return __builtin_shufflevector(lo, hi, 0, 1, 2, 3, 4, 5, 6, 7,
                                 8, 9, 10, 11, 12, 13, 14, 15);
}

// ---------------------------------------------------------------------------
// WMMA macro-tile engine: 32x64 f32 accum (2x4 grid of 16x16 WMMA tiles),
// NT only: A row-major [M,Kd], B row-major [N,Kd], C = A.B^T.
// 8 independent accumulators (64 VGPR) + 6 frags (48 VGPR): no spills,
// 1.5 b128 loads per WMMA. All fragment loads are contiguous 16B vectors
// (contiguous k = 16h + s order, same for A and B: chunks at +16h/+16h+8).
// ---------------------------------------------------------------------------
__device__ inline void wmma_block(const __bf16* A, int lda,
                                  const __bf16* B, int ldb,
                                  int row0, int col0, int kdim, v8f c[8]) {
  const int lane = threadIdx.x & 31;
  const int q = lane & 15, h = lane >> 4;
  const __bf16* pa0 = A + (size_t)(row0 + q) * lda + (h << 4);
  const __bf16* pa1 = pa0 + (size_t)16 * lda;
  const __bf16* pb0 = B + (size_t)(col0 + q) * ldb + (h << 4);
  const __bf16* pb1 = pb0 + (size_t)16 * ldb;
  const __bf16* pb2 = pb0 + (size_t)32 * ldb;
  const __bf16* pb3 = pb0 + (size_t)48 * ldb;
  v8f z = {};
#pragma unroll
  for (int i = 0; i < 8; ++i) c[i] = z;
  for (int k0 = 0; k0 < kdim; k0 += 32) {
    v16bf a0 = cat8(*(const v8bf*)(pa0 + k0), *(const v8bf*)(pa0 + k0 + 8));
    v16bf a1 = cat8(*(const v8bf*)(pa1 + k0), *(const v8bf*)(pa1 + k0 + 8));
    v16bf b0 = cat8(*(const v8bf*)(pb0 + k0), *(const v8bf*)(pb0 + k0 + 8));
    v16bf b1 = cat8(*(const v8bf*)(pb1 + k0), *(const v8bf*)(pb1 + k0 + 8));
    v16bf b2 = cat8(*(const v8bf*)(pb2 + k0), *(const v8bf*)(pb2 + k0 + 8));
    v16bf b3 = cat8(*(const v8bf*)(pb3 + k0), *(const v8bf*)(pb3 + k0 + 8));
    c[0] = __builtin_amdgcn_wmma_f32_16x16x32_bf16(false, a0, false, b0, (short)0, c[0], false, false);
    c[1] = __builtin_amdgcn_wmma_f32_16x16x32_bf16(false, a0, false, b1, (short)0, c[1], false, false);
    c[2] = __builtin_amdgcn_wmma_f32_16x16x32_bf16(false, a0, false, b2, (short)0, c[2], false, false);
    c[3] = __builtin_amdgcn_wmma_f32_16x16x32_bf16(false, a0, false, b3, (short)0, c[3], false, false);
    c[4] = __builtin_amdgcn_wmma_f32_16x16x32_bf16(false, a1, false, b0, (short)0, c[4], false, false);
    c[5] = __builtin_amdgcn_wmma_f32_16x16x32_bf16(false, a1, false, b1, (short)0, c[5], false, false);
    c[6] = __builtin_amdgcn_wmma_f32_16x16x32_bf16(false, a1, false, b2, (short)0, c[6], false, false);
    c[7] = __builtin_amdgcn_wmma_f32_16x16x32_bf16(false, a1, false, b3, (short)0, c[7], false, false);
  }
}

__device__ inline void store_block_f32(float* C, int ldc, int row0, int col0,
                                       const v8f c[8], float scale) {
  int lane = threadIdx.x & 31, n0 = lane & 15, h = lane >> 4;
#pragma unroll
  for (int u = 0; u < 2; ++u)
#pragma unroll
    for (int v = 0; v < 4; ++v)
#pragma unroll
      for (int j = 0; j < 8; ++j) {
        int m = row0 + 16 * u + j + (h << 3);
        int n = col0 + 16 * v + n0;
        C[(size_t)m * ldc + n] = c[u * 4 + v][j] * scale;
      }
}
__device__ inline void store_block_bf16(__bf16* C, int ldc, int row0, int col0,
                                        const v8f c[8]) {
  int lane = threadIdx.x & 31, n0 = lane & 15, h = lane >> 4;
#pragma unroll
  for (int u = 0; u < 2; ++u)
#pragma unroll
    for (int v = 0; v < 4; ++v)
#pragma unroll
      for (int j = 0; j < 8; ++j) {
        int m = row0 + 16 * u + j + (h << 3);
        int n = col0 + 16 * v + n0;
        C[(size_t)m * ldc + n] = (__bf16)c[u * 4 + v][j];
      }
}

// ---------------------------------------------------------------------------
// Conversion / transpose kernels
// ---------------------------------------------------------------------------
__global__ void k_cvt_bf16(const float* __restrict__ src, __bf16* __restrict__ dst,
                           size_t n) {
  for (size_t i = blockIdx.x * (size_t)blockDim.x + threadIdx.x; i < n;
       i += (size_t)gridDim.x * blockDim.x)
    dst[i] = (__bf16)src[i];
}
// Wn_w[:,1:] -> bf16 [K,D]
__global__ void k_cvt_wn(const float* __restrict__ Wn_w, __bf16* __restrict__ Wn2h) {
  for (size_t i = blockIdx.x * (size_t)blockDim.x + threadIdx.x; i < nKD;
       i += (size_t)gridDim.x * blockDim.x) {
    size_t k = i / cD, d = i - k * cD;
    Wn2h[i] = (__bf16)Wn_w[k * (cD + 1) + 1 + d];
  }
}
// Extract inverse (right half of augmented) -> bf16 [K,K] (symmetric)
__global__ void k_cvt_inv(const float* __restrict__ Aug, __bf16* __restrict__ invh) {
  for (size_t i = blockIdx.x * (size_t)blockDim.x + threadIdx.x; i < nKK;
       i += (size_t)gridDim.x * blockDim.x) {
    size_t r = i / cK, c = i - r * cK;
    invh[i] = (__bf16)Aug[r * (2 * cK) + cK + c];
  }
}
// V^T per batch: VhT[b][d][t] = V[b][t][d]  (f32 -> bf16)
__global__ void k_transpose_vt(const float* __restrict__ Vf, __bf16* __restrict__ VhT) {
  for (size_t i = blockIdx.x * (size_t)blockDim.x + threadIdx.x; i < nBTD;
       i += (size_t)gridDim.x * blockDim.x) {
    size_t b = i / ((size_t)cT * cD);
    size_t rem = i - b * (size_t)cT * cD;
    size_t d = rem / cT, t = rem - d * cT;
    VhT[i] = (__bf16)Vf[(b * cT + t) * cD + d];
  }
}
// S^T: ShT[d][k] = S_init[k][d]  (f32 -> bf16)
__global__ void k_transpose_st(const float* __restrict__ S, __bf16* __restrict__ ShT) {
  for (size_t i = blockIdx.x * (size_t)blockDim.x + threadIdx.x; i < nKD;
       i += (size_t)gridDim.x * blockDim.x) {
    size_t d = i / cK, k = i - d * cK;
    ShT[i] = (__bf16)S[k * cD + d];
  }
}
// Row-normalize (eps-clamped) f32 -> bf16; one wave per row.
__global__ void k_rownorm(const float* __restrict__ src, __bf16* __restrict__ dst,
                          int rows, int cols) {
  int wave = threadIdx.x >> 5, lane = threadIdx.x & 31;
  int r = blockIdx.x * (blockDim.x >> 5) + wave;
  if (r >= rows) return;
  const float* s = src + (size_t)r * cols;
  float ss = 0.f;
  for (int c = lane; c < cols; c += 32) { float v = s[c]; ss += v * v; }
  ss = wave_sum(ss);
  float inv = 1.0f / fmaxf(sqrtf(ss), 1e-8f);
  __bf16* d = dst + (size_t)r * cols;
  for (int c = lane; c < cols; c += 32) d[c] = (__bf16)(s[c] * inv);
}

// ---------------------------------------------------------------------------
// GEMM kernels (all NT, 32x64-per-wave WMMA macro-tiles)
// ---------------------------------------------------------------------------
// Q/V projection: C = A(bf16).W^T + bias, f32 out (+ optional bf16 copy).
__global__ void k_gemm_qv(const __bf16* __restrict__ A, const __bf16* __restrict__ W,
                          const float* __restrict__ bias,
                          float* __restrict__ Cf, __bf16* __restrict__ Ch) {
  int row0, col0;
  if (!tile_coords((int)nBT, cD, row0, col0)) return;
  v8f c[8];
  wmma_block(A, cD, W, cD, row0, col0, cD, c);
  int lane = threadIdx.x & 31, n0 = lane & 15, h = lane >> 4;
#pragma unroll
  for (int u = 0; u < 2; ++u)
#pragma unroll
    for (int v = 0; v < 4; ++v)
#pragma unroll
      for (int j = 0; j < 8; ++j) {
        int m = row0 + 16 * u + j + (h << 3);
        int n = col0 + 16 * v + n0;
        float val = c[u * 4 + v][j] + bias[n];
        size_t off = (size_t)m * cD + n;
        Cf[off] = val;
        if (Ch) Ch[off] = (__bf16)val;
      }
}

// Generic NT GEMM, batched over blockIdx.y, optional causal tile skip.
__global__ void k_gemm_nt(const __bf16* __restrict__ A, size_t sA,
                          const __bf16* __restrict__ B, size_t sB,
                          float* __restrict__ C, size_t sC,
                          int M, int N, int Kd, float scale, int causal) {
  int row0, col0;
  if (!tile_coords(M, N, row0, col0)) return;
  if (causal && col0 > row0 + 31) return;
  int b = blockIdx.y;
  v8f c[8];
  wmma_block(A + (size_t)b * sA, Kd, B + (size_t)b * sB, Kd, row0, col0, Kd, c);
  store_block_f32(C + (size_t)b * sC, N, row0, col0, c, scale);
}

// bypass = attn(bf16) . V : NT against V^T [D,T], causal depth limit.
__global__ void k_gemm_bypass(const __bf16* __restrict__ attn,
                              const __bf16* __restrict__ VhT,
                              float* __restrict__ byp) {
  int row0, col0;
  if (!tile_coords(cT, cD, row0, col0)) return;
  int b = blockIdx.y;
  int kd = row0 + 32; if (kd > cT) kd = cT;   // attn is zero beyond diagonal
  v8f c[8];
  wmma_block(attn + (size_t)b * cT * cT, cT,
             VhT + (size_t)b * cT * cD, cT, row0, col0, kd, c);
  store_block_f32(byp + (size_t)b * cT * cD, cD, row0, col0, c, 1.0f);
}

// V_agg = rwT . V^T^T : NT with A = rwT [K,T], B = V^T [D,T]; dual output.
__global__ void k_gemm_vagg(const __bf16* __restrict__ rwT,
                            const __bf16* __restrict__ VhT,
                            float* __restrict__ Vaggf, __bf16* __restrict__ Vaggh) {
  int row0, col0;
  if (!tile_coords(cK, cD, row0, col0)) return;
  int b = blockIdx.y;
  v8f c[8];
  wmma_block(rwT + (size_t)b * cT * cK, cT,
             VhT + (size_t)b * cT * cD, cT, row0, col0, cT, c);
  int lane = threadIdx.x & 31, n0 = lane & 15, h = lane >> 4;
#pragma unroll
  for (int u = 0; u < 2; ++u)
#pragma unroll
    for (int v = 0; v < 4; ++v)
#pragma unroll
      for (int j = 0; j < 8; ++j) {
        int m = row0 + 16 * u + j + (h << 3);
        int n = col0 + 16 * v + n0;
        size_t off = (size_t)b * nKD + (size_t)m * cD + n;
        float val = c[u * 4 + v][j];
        Vaggf[off] = val; Vaggh[off] = (__bf16)val;
      }
}

// S_cov = S.S^T + 1e-5 I written into left half of augmented [K, 2K].
__global__ void k_gemm_scov(const __bf16* __restrict__ Sh, float* __restrict__ Aug) {
  int row0, col0;
  if (!tile_coords(cK, cK, row0, col0)) return;
  v8f c[8];
  wmma_block(Sh, cD, Sh, cD, row0, col0, cD, c);
  int lane = threadIdx.x & 31, n0 = lane & 15, h = lane >> 4;
#pragma unroll
  for (int u = 0; u < 2; ++u)
#pragma unroll
    for (int v = 0; v < 4; ++v)
#pragma unroll
      for (int j = 0; j < 8; ++j) {
        int m = row0 + 16 * u + j + (h << 3);
        int n = col0 + 16 * v + n0;
        Aug[(size_t)m * (2 * cK) + n] = c[u * 4 + v][j] + (m == n ? 1e-5f : 0.0f);
      }
}

// SV = V_agg.S^T (NT), bf16 out, batched.
__global__ void k_gemm_sv(const __bf16* __restrict__ Vaggh,
                          const __bf16* __restrict__ Sh, __bf16* __restrict__ SVh) {
  int row0, col0;
  if (!tile_coords(cK, cK, row0, col0)) return;
  int b = blockIdx.y;
  v8f c[8];
  wmma_block(Vaggh + (size_t)b * nKD, cD, Sh, cD, row0, col0, cD, c);
  store_block_bf16(SVh + (size_t)b * nKK, cK, row0, col0, c);
}

// proj = SV . S_cov_inv : inverse is symmetric (SPD), so NT with invh is exact.
__global__ void k_gemm_proj(const __bf16* __restrict__ SVh,
                            const __bf16* __restrict__ invh,
                            __bf16* __restrict__ projh) {
  int row0, col0;
  if (!tile_coords(cK, cK, row0, col0)) return;
  int b = blockIdx.y;
  v8f c[8];
  wmma_block(SVh + (size_t)b * nKK, cK, invh, cK, row0, col0, cK, c);
  store_block_bf16(projh + (size_t)b * nKK, cK, row0, col0, c);
}

// S_next = (1-LAM) S + ETA (V_agg - proj.S) : NT with B = S^T [D,K].
__global__ void k_gemm_snext(const __bf16* __restrict__ projh,
                             const __bf16* __restrict__ ShT,
                             const float* __restrict__ Vaggf,
                             const float* __restrict__ S_init,
                             float* __restrict__ outS) {
  int row0, col0;
  if (!tile_coords(cK, cD, row0, col0)) return;
  int b = blockIdx.y;
  v8f c[8];
  wmma_block(projh + (size_t)b * nKK, cK, ShT, cK, row0, col0, cK, c);
  int lane = threadIdx.x & 31, n0 = lane & 15, h = lane >> 4;
#pragma unroll
  for (int u = 0; u < 2; ++u)
#pragma unroll
    for (int v = 0; v < 4; ++v)
#pragma unroll
      for (int j = 0; j < 8; ++j) {
        int m = row0 + 16 * u + j + (h << 3);
        int n = col0 + 16 * v + n0;
        size_t off = (size_t)m * cD + n;
        outS[(size_t)b * nKD + off] =
            (1.0f - cLAM) * S_init[off] +
            cETA * (Vaggf[(size_t)b * nKD + off] - c[u * 4 + v][j]);
      }
}

// ---------------------------------------------------------------------------
// Row-wise / pointwise kernels
// ---------------------------------------------------------------------------
// Per (b,t): top-16 of read_logits row, softmax, context = sum p_i * S[idx_i].
__global__ void k_topk_context(const float* __restrict__ logits,
                               const float* __restrict__ S,
                               float* __restrict__ ctx) {
  int wave = threadIdx.x >> 5, lane = threadIdx.x & 31;
  int r = blockIdx.x * (blockDim.x >> 5) + wave;
  if (r >= (int)nBT) return;
  const float* row = logits + (size_t)r * cK;
  float lv[16];
#pragma unroll
  for (int i = 0; i < 16; ++i) lv[i] = row[lane + 32 * i];
  float selv[cKR]; int seli[cKR];
  for (int s = 0; s < cKR; ++s) {
    float m = -3.0e38f; int mi = 0;
#pragma unroll
    for (int i = 0; i < 16; ++i)
      if (lv[i] > m) { m = lv[i]; mi = lane + 32 * i; }
#pragma unroll
    for (int o = 16; o > 0; o >>= 1) {
      float om = __shfl_xor(m, o, 32);
      int   oi = __shfl_xor(mi, o, 32);
      if (om > m || (om == m && oi < mi)) { m = om; mi = oi; }
    }
    selv[s] = m; seli[s] = mi;
    if ((mi & 31) == lane) {
      int ii = mi >> 5;
#pragma unroll
      for (int i = 0; i < 16; ++i) if (i == ii) lv[i] = -3.0e38f;
    }
  }
  float p[cKR]; float ssum = 0.f;
#pragma unroll
  for (int s = 0; s < cKR; ++s) { p[s] = __expf(selv[s] - selv[0]); ssum += p[s]; }
  float inv = 1.0f / ssum;
#pragma unroll
  for (int s = 0; s < cKR; ++s) p[s] *= inv;
  for (int d = lane; d < cD; d += 32) {
    float acc = 0.f;
#pragma unroll
    for (int s = 0; s < cKR; ++s) acc += p[s] * S[(size_t)seli[s] * cD + d];
    ctx[(size_t)r * cD + d] = acc;
  }
}

// Causal row softmax, f32 scores -> bf16 attn (zeros beyond the diagonal).
__global__ void k_softmax_causal(const float* __restrict__ scores,
                                 __bf16* __restrict__ attn) {
  int wave = threadIdx.x >> 5, lane = threadIdx.x & 31;
  int r = blockIdx.x * (blockDim.x >> 5) + wave;
  if (r >= (int)nBT) return;
  int q = r & (cT - 1);
  const float* row = scores + (size_t)r * cT;
  __bf16* arow = attn + (size_t)r * cT;
  int len = q + 1;
  float m = -3.0e38f;
  for (int k = lane; k < len; k += 32) m = fmaxf(m, row[k]);
  m = wave_max(m);
  float s = 0.f;
  for (int k = lane; k < len; k += 32) s += __expf(row[k] - m);
  s = wave_sum(s);
  float inv = 1.0f / s;
  for (int k = lane; k < len; k += 32) arow[k] = (__bf16)(__expf(row[k] - m) * inv);
  for (int k = len + lane; k < cT; k += 32) arow[k] = (__bf16)0.0f;
}

// instr_prob = sigmoid(Q.Wm + b); one wave per row.
__global__ void k_instr_prob(const float* __restrict__ Qf,
                             const float* __restrict__ Wm_w,
                             const float* __restrict__ Wm_b,
                             float* __restrict__ prob) {
  int wave = threadIdx.x >> 5, lane = threadIdx.x & 31;
  int r = blockIdx.x * (blockDim.x >> 5) + wave;
  if (r >= (int)nBT) return;
  const float* q = Qf + (size_t)r * cD;
  float s = 0.f;
  for (int d = lane; d < cD; d += 32) s += q[d] * Wm_w[d];
  s = wave_sum(s);
  if (lane == 0) prob[r] = 1.0f / (1.0f + __expf(-(s + Wm_b[0])));
}

// out = (1-p)*ctx(out) + p*byp
__global__ void k_mix(float* __restrict__ out, const float* __restrict__ byp,
                      const float* __restrict__ prob) {
  for (size_t i = blockIdx.x * (size_t)blockDim.x + threadIdx.x; i < nBTD;
       i += (size_t)gridDim.x * blockDim.x) {
    float p = prob[i / cD];
    out[i] = (1.0f - p) * out[i] + p * byp[i];
  }
}

// novelty = 1 - rowmax(sim); one wave per row over K=512.
__global__ void k_novelty(const float* __restrict__ sim, float* __restrict__ nov) {
  int wave = threadIdx.x >> 5, lane = threadIdx.x & 31;
  int r = blockIdx.x * (blockDim.x >> 5) + wave;
  if (r >= (int)nBT) return;
  const float* row = sim + (size_t)r * cK;
  float m = -3.0e38f;
  for (int k = lane; k < cK; k += 32) m = fmaxf(m, row[k]);
  m = wave_max(m);
  if (lane == 0) nov[r] = 1.0f - m;
}

// r_write epilogue: sigmoid(logits + Wn_b + nov*Wn_col0) * w[t], written
// TRANSPOSED as rwT [B][K][T] (bf16) so V_agg becomes a pure NT GEMM.
__global__ void k_rwrite_epi(const float* __restrict__ logits,
                             const float* __restrict__ nov,
                             const float* __restrict__ Wn_w,
                             const float* __restrict__ Wn_b,
                             __bf16* __restrict__ rwT) {
  float wnorm = cLAM / (1.0f - __powf(1.0f - cLAM, (float)cT));
  for (size_t i = blockIdx.x * (size_t)blockDim.x + threadIdx.x; i < nBTK;
       i += (size_t)gridDim.x * blockDim.x) {
    size_t r = i / cK, k = i - r * cK;
    size_t b = r / cT;
    int t = (int)(r & (cT - 1));
    float xv = logits[i] + Wn_b[k] + nov[r] * Wn_w[k * (cD + 1)];
    float sg = 1.0f / (1.0f + __expf(-xv));
    float wt = __powf(1.0f - cLAM, (float)(cT - 1 - t)) * wnorm;
    rwT[(b * cK + k) * cT + t] = (__bf16)(sg * wt);
  }
}

// Identity into right half of augmented matrix.
__global__ void k_init_identity(float* __restrict__ Aug) {
  for (size_t i = blockIdx.x * (size_t)blockDim.x + threadIdx.x; i < nKK;
       i += (size_t)gridDim.x * blockDim.x) {
    size_t r = i / cK, c = i - r * cK;
    Aug[r * (2 * cK) + cK + c] = (r == c) ? 1.0f : 0.0f;
  }
}

// Single-block Gauss-Jordan on SPD 512x512 augmented [512][1024].
__global__ void __launch_bounds__(1024) k_gauss(float* __restrict__ Aug) {
  const int W = 2 * cK;
  const int tid = threadIdx.x;           // one column per thread (blockDim==1024)
  __shared__ float prow[2 * cK];
  __shared__ float colp[cK];
  for (int p = 0; p < cK; ++p) {
    prow[tid] = Aug[(size_t)p * W + tid];
    __syncthreads();
    float scale = 1.0f / prow[p];
    __syncthreads();
    prow[tid] *= scale;
    if (tid < cK) colp[tid] = (tid == p) ? 0.0f : Aug[(size_t)tid * W + p];
    __syncthreads();
    Aug[(size_t)p * W + tid] = prow[tid];
    float pv = prow[tid];
    for (int r = 0; r < cK; ++r) {
      float f = colp[r];
      if (f != 0.0f) Aug[(size_t)r * W + tid] -= f * pv;
    }
    __syncthreads();
  }
}

// ---------------------------------------------------------------------------
// Host launch
// ---------------------------------------------------------------------------
extern "C" void kernel_launch(void* const* d_in, const int* in_sizes, int n_in,
                              void* d_out, int out_size, void* d_ws, size_t ws_size,
                              hipStream_t stream) {
  (void)in_sizes; (void)n_in; (void)out_size; (void)ws_size;
  const float* x      = (const float*)d_in[0];
  const float* S_init = (const float*)d_in[1];
  const float* Wq_w   = (const float*)d_in[2];
  const float* Wq_b   = (const float*)d_in[3];
  const float* Wv_w   = (const float*)d_in[4];
  const float* Wv_b   = (const float*)d_in[5];
  const float* Wn_w   = (const float*)d_in[6];
  const float* Wn_b   = (const float*)d_in[7];
  const float* Wm_w   = (const float*)d_in[8];
  const float* Wm_b   = (const float*)d_in[9];
  float* out  = (float*)d_out;             // [B,T,D]
  float* outS = out + nBTD;                // [B,K,D]

  char* wp = (char*)d_ws;
  auto alloc = [&](size_t bytes) -> void* {
    char* r = wp; wp += (bytes + 255) & ~(size_t)255; return (void*)r;
  };
  __bf16* xh    = (__bf16*)alloc(nBTD * 2);   // later reused as Vnh
  __bf16* Wqh   = (__bf16*)alloc(nDD * 2);
  __bf16* Wvh   = (__bf16*)alloc(nDD * 2);
  __bf16* Wn2h  = (__bf16*)alloc(nKD * 2);
  __bf16* Sh    = (__bf16*)alloc(nKD * 2);
  __bf16* Snh   = (__bf16*)alloc(nKD * 2);
  __bf16* ShT   = (__bf16*)alloc(nKD * 2);
  float*  Qf    = (float*)alloc(nBTD * 4);
  float*  Vf    = (float*)alloc(nBTD * 4);
  __bf16* Qh    = (__bf16*)alloc(nBTD * 2);
  __bf16* VhT   = (__bf16*)alloc(nBTD * 2);   // V^T per batch [B][D][T]
  float*  btk   = (float*)alloc(nBTK * 4);    // read_logits -> sim -> rw logits
  float*  scores= (float*)alloc(nBTT * 4);    // later reused as bypass (smaller)
  __bf16* attn  = (__bf16*)alloc(nBTT * 2);
  float*  prob  = (float*)alloc(nBT * 4);
  float*  nov   = (float*)alloc(nBT * 4);
  __bf16* rwT   = (__bf16*)alloc(nBTK * 2);   // write gate, transposed [B][K][T]
  float*  Vaggf = (float*)alloc(nBKD * 4);
  __bf16* Vaggh = (__bf16*)alloc(nBKD * 2);
  float*  Aug   = (float*)alloc((size_t)cK * 2 * cK * 4);
  __bf16* invh  = (__bf16*)alloc(nKK * 2);
  __bf16* SVh   = (__bf16*)alloc(nBKK * 2);
  __bf16* projh = (__bf16*)alloc(nBKK * 2);
  __bf16* Vnh = xh;        // alias: xh dead after Q/V projections
  float*  byp = scores;    // alias: scores dead after softmax

  auto gb = [](int M, int N) { return ((M / 32) * (N / 64) + 7) / 8; };
  auto eb = [](size_t n) { int b = (int)((n + 255) / 256); return b > 8192 ? 8192 : b; };
  const float invSqrtD = 1.0f / 32.0f;     // 1/sqrt(1024)

  // 1) conversions
  k_cvt_bf16<<<eb(nBTD), 256, 0, stream>>>(x, xh, nBTD);
  k_cvt_bf16<<<eb(nDD), 256, 0, stream>>>(Wq_w, Wqh, nDD);
  k_cvt_bf16<<<eb(nDD), 256, 0, stream>>>(Wv_w, Wvh, nDD);
  k_cvt_bf16<<<eb(nKD), 256, 0, stream>>>(S_init, Sh, nKD);
  k_cvt_wn<<<eb(nKD), 256, 0, stream>>>(Wn_w, Wn2h);
  k_transpose_st<<<eb(nKD), 256, 0, stream>>>(S_init, ShT);

  // 2) Q/V projections (WMMA)
  k_gemm_qv<<<gb(nBT, cD), 256, 0, stream>>>(xh, Wqh, Wq_b, Qf, Qh);
  k_gemm_qv<<<gb(nBT, cD), 256, 0, stream>>>(xh, Wvh, Wv_b, Vf, (__bf16*)nullptr);
  k_transpose_vt<<<eb(nBTD), 256, 0, stream>>>(Vf, VhT);

  // 3) sparse read: logits, top-k, context (context into d_out)
  k_gemm_nt<<<gb(nBT, cK), 256, 0, stream>>>(Qh, 0, Sh, 0, btk, 0,
                                             (int)nBT, cK, cD, invSqrtD, 0);
  k_topk_context<<<(int)(nBT / 8), 256, 0, stream>>>(btk, S_init, out);

  // 4) novelty: normalized rows, sim GEMM, row-max
  k_rownorm<<<(int)(nBT / 8), 256, 0, stream>>>(Vf, Vnh, (int)nBT, cD);
  k_rownorm<<<cK / 8, 256, 0, stream>>>(S_init, Snh, cK, cD);
  k_gemm_nt<<<gb(nBT, cK), 256, 0, stream>>>(Vnh, 0, Snh, 0, btk, 0,
                                             (int)nBT, cK, cD, 1.0f, 0);
  k_novelty<<<(int)(nBT / 8), 256, 0, stream>>>(btk, nov);

  // 5) write gate: GEMM over Q part, epilogue adds novelty rank-1 + bias,
  //    folds sigmoid + temporal decay w[t], stores transposed [B][K][T]
  k_gemm_nt<<<gb(nBT, cK), 256, 0, stream>>>(Qh, 0, Wn2h, 0, btk, 0,
                                             (int)nBT, cK, cD, 1.0f, 0);
  k_rwrite_epi<<<eb(nBTK), 256, 0, stream>>>(btk, nov, Wn_w, Wn_b, rwT);

  // 6) causal self-attention bypass
  k_gemm_nt<<<dim3(gb(cT, cT), cB), 256, 0, stream>>>(
      Qh, (size_t)cT * cD, Qh, (size_t)cT * cD, scores, (size_t)cT * cT,
      cT, cT, cD, invSqrtD, 1);
  k_softmax_causal<<<(int)(nBT / 8), 256, 0, stream>>>(scores, attn);
  k_gemm_bypass<<<dim3(gb(cT, cD), cB), 256, 0, stream>>>(attn, VhT, byp);
  k_instr_prob<<<(int)(nBT / 8), 256, 0, stream>>>(Qf, Wm_w, Wm_b, prob);
  k_mix<<<eb(nBTD), 256, 0, stream>>>(out, byp, prob);

  // 7) V_agg = rwT . V^T (NT WMMA)
  k_gemm_vagg<<<dim3(gb(cK, cD), cB), 256, 0, stream>>>(rwT, VhT, Vaggf, Vaggh);

  // 8) S_cov + inverse (batch-independent: S_prev broadcast of S_init)
  k_init_identity<<<eb(nKK), 256, 0, stream>>>(Aug);
  k_gemm_scov<<<gb(cK, cK), 256, 0, stream>>>(Sh, Aug);
  k_gauss<<<1, 1024, 0, stream>>>(Aug);
  k_cvt_inv<<<eb(nKK), 256, 0, stream>>>(Aug, invh);

  // 9) orthogonal projection + state update (into d_out tail)
  k_gemm_sv<<<dim3(gb(cK, cK), cB), 256, 0, stream>>>(Vaggh, Sh, SVh);
  k_gemm_proj<<<dim3(gb(cK, cK), cB), 256, 0, stream>>>(SVh, invh, projh);
  k_gemm_snext<<<dim3(gb(cK, cD), cB), 256, 0, stream>>>(projh, ShT, Vaggf,
                                                         S_init, outS);
}
